// MultiHeadAttention_31147102831139
// MI455X (gfx1250) — compile-verified, matrix operand order checked by reading, not run on hardware
//
#include <hip/hip_runtime.h>
#include <hip/hip_bf16.h>

#define DEV __device__ __forceinline__

typedef __attribute__((ext_vector_type(16))) _Float16 v16h;
typedef __attribute__((ext_vector_type(8)))  _Float16 v8h;
typedef __attribute__((ext_vector_type(8)))  float    v8f;
typedef __attribute__((ext_vector_type(4)))  float    v4f;
typedef __attribute__((ext_vector_type(4)))  int      i32x4;
typedef __attribute__((ext_vector_type(8)))  int      i32x8;
typedef __attribute__((ext_vector_type(4)))  unsigned int u32x4;

static constexpr int Bsz = 4;
static constexpr int Sq  = 2048;
static constexpr int Dm  = 1024;
static constexpr int Hh  = 16;
static constexpr int Dh  = 64;
#define NEGF (-1.0e9f)
#define AWAVES 8   // waves per attention block
#define QROWS 32   // q rows per wave (2 row-groups of 16)

#if __has_builtin(__builtin_amdgcn_global_load_async_to_lds_b128) && \
    __has_builtin(__builtin_amdgcn_s_wait_asynccnt)
#define HAVE_ASYNC_LDS 1
#else
#define HAVE_ASYNC_LDS 0
#endif

#if __has_builtin(__builtin_amdgcn_tensor_load_to_lds) && \
    __has_builtin(__builtin_amdgcn_s_wait_tensorcnt)
#define HAVE_TDM 1
#else
#define HAVE_TDM 0
#endif

#if __has_include(<hip/amd_detail/amd_gfx1250_TDM.h>)
#define TDM_SIX_ARGS 1   // therock headers -> 6-arg builtin
#else
#define TDM_SIX_ARGS 0   // ROCm 7.2 -> 5-arg builtin
#endif

typedef __attribute__((address_space(1))) i32x4 as1_i32x4;
typedef __attribute__((address_space(3))) i32x4 as3_i32x4;

// ---------------------------------------------------------------------------
// D = A(16x32 f16) * B(32x16 f16) + C(16x16 f32)
DEV v8f wmma_f16(v16h a, v16h b, v8f c) {
  return __builtin_amdgcn_wmma_f32_16x16x32_f16(
      /*neg_a=*/false, a, /*neg_b=*/false, b,
      /*c_mod=*/(short)0, c, /*reuse_a=*/false, /*reuse_b=*/false);
}

// A-matrix (16x32, M x K) load from row-major f16 [M][K] with leading dim ld.
// Same pattern serves B-matrix (32x16, K x N) when source is stored [N][K].
// Lane L<16: M(N)=L, K chunks {0..7,16..23}; lanes 16..31: {8..15,24..31}.
DEV v16h load_tile_f16(const _Float16* p, int ld, int lane) {
  const _Float16* base = p + (size_t)(lane & 15) * ld + ((lane >> 4) << 3);
  v8h c0 = *(const v8h*)(base);
  v8h c1 = *(const v8h*)(base + 16);
  v16h r;
#pragma unroll
  for (int j = 0; j < 8; ++j) { r[j] = c0[j]; r[j + 8] = c1[j]; }
  return r;
}

// Same tile pattern but f32 source, converted to f16 on the fly.
DEV v16h load_tile_f32(const float* p, int ld, int lane) {
  const float* base = p + (size_t)(lane & 15) * ld + ((lane >> 4) << 3);
  v4f a0 = *(const v4f*)(base);
  v4f a1 = *(const v4f*)(base + 4);
  v4f b0 = *(const v4f*)(base + 16);
  v4f b1 = *(const v4f*)(base + 20);
  v16h r;
#pragma unroll
  for (int j = 0; j < 4; ++j) {
    r[j]      = (_Float16)a0[j];
    r[j + 4]  = (_Float16)a1[j];
    r[j + 8]  = (_Float16)b0[j];
    r[j + 12] = (_Float16)b1[j];
  }
  return r;
}

// 16-byte global->LDS copy: async DMA (ASYNCcnt) when available.
DEV void copy16_g2l(const _Float16* g, _Float16* l) {
#if HAVE_ASYNC_LDS
  __builtin_amdgcn_global_load_async_to_lds_b128(
      (as1_i32x4*)(void*)const_cast<_Float16*>(g),
      (as3_i32x4*)(void*)l,
      /*offset=*/0, /*cpol=*/0);
#else
  *(v8h*)l = *(const v8h*)g;
#endif
}

DEV void wait_async_copies() {
#if HAVE_ASYNC_LDS
  __builtin_amdgcn_s_wait_asynccnt(0);
#endif
}

#if HAVE_TDM
// TDM: DMA a 2D f16 tile (tile_w elems x tile_h rows, row stride in elems)
// from global memory into LDS (packed row-major, ld = tile_w).
// D# per CDNA5 ISA ch.8: group0 = {count, lds_addr, global_addr, type=2},
// group1 = {data_size=2B, tensor dims, tile dims, dim0 stride}. 2D -> groups
// 2/3 zero. Issued by one wave; tracked by TENSORcnt.
DEV void tdm_load_2d(const _Float16* gsrc, _Float16* ldst,
                     unsigned tile_w, unsigned tile_h, unsigned row_stride) {
  const unsigned long long ga = (unsigned long long)(size_t)gsrc;
  const unsigned lds = (unsigned)(size_t)ldst;   // low 32 bits = LDS byte addr
  u32x4 g0;
  g0[0] = 1u;                                        // count=1, user D#
  g0[1] = lds;                                       // lds_addr
  g0[2] = (unsigned)(ga & 0xffffffffu);              // global_addr[31:0]
  g0[3] = (unsigned)((ga >> 32) & 0x1ffffffu)        // global_addr[56:32]
        | (2u << 30);                                // type = 2 ("image")
  i32x8 g1 = {};
  g1[0] = (int)(1u << 16);                           // data_size = 1 -> 2 bytes
  g1[1] = (int)((tile_w & 0xffffu) << 16);           // tensor_dim0[15:0]
  g1[2] = (int)(((tile_w >> 16) & 0xffffu)           // tensor_dim0[31:16]
        |       ((tile_h & 0xffffu) << 16));         // tensor_dim1[15:0]
  g1[3] = (int)(((tile_h >> 16) & 0xffffu)           // tensor_dim1[31:16]
        |       ((tile_w & 0xffffu) << 16));         // tile_dim0
  g1[4] = (int)(tile_h & 0xffffu);                   // tile_dim1 (tile_dim2=0)
  g1[5] = (int)row_stride;                           // tensor_dim0_stride[31:0]
  g1[6] = 0;                                         // stride[47:32], dim1_stride lo
  g1[7] = 0;
  i32x4 gz = {};
#if TDM_SIX_ARGS
  i32x8 gz8 = {};
  __builtin_amdgcn_tensor_load_to_lds(g0, g1, gz, gz, gz8, 0);
#else
  __builtin_amdgcn_tensor_load_to_lds(g0, g1, gz, gz, 0);
#endif
}
#endif  // HAVE_TDM

// ---------------------------------------------------------------------------
// Kernel 1: transpose + convert weights (1024x1024 f32 [k][n]) -> f16 W^T [n][k].
__global__ __launch_bounds__(256) void wtrans_kernel(
    const float* __restrict__ wq, const float* __restrict__ wk,
    const float* __restrict__ wv, _Float16* __restrict__ tq,
    _Float16* __restrict__ tk, _Float16* __restrict__ tv) {
  const float* w = (blockIdx.y == 0) ? wq : (blockIdx.y == 1) ? wk : wv;
  _Float16* o    = (blockIdx.y == 0) ? tq : (blockIdx.y == 1) ? tk : tv;
  int idx = blockIdx.x * 256 + threadIdx.x;   // k*1024 + n (coalesced read)
  int n = idx & (Dm - 1);
  int kk = idx >> 10;
  o[(size_t)n * Dm + kk] = (_Float16)w[idx];
}

// ---------------------------------------------------------------------------
// Kernel 2: fused QKV projection GEMM (wave32 WMMA).
// grid.x = (B*S)/16 row tiles, grid.y = 3 (q,k,v). Block = 256 (8 waves);
// wave w computes rows[16] x cols[w*128, w*128+128) over K=1024.
// Q,K heads -> [b][h][s][64] f16; V -> transposed [b][h][d][s] f16.
__global__ __launch_bounds__(256) void proj_kernel(
    const float* __restrict__ xq, const float* __restrict__ xk,
    const float* __restrict__ xv, const _Float16* __restrict__ wtq,
    const _Float16* __restrict__ wtk, const _Float16* __restrict__ wtv,
    const float* __restrict__ bq, const float* __restrict__ bk,
    const float* __restrict__ bv, _Float16* __restrict__ qh,
    _Float16* __restrict__ kh, _Float16* __restrict__ vt) {
  const int which = blockIdx.y;
  const float* x      = (which == 0) ? xq  : (which == 1) ? xk  : xv;
  const _Float16* wt  = (which == 0) ? wtq : (which == 1) ? wtk : wtv;
  const float* bias   = (which == 0) ? bq  : (which == 1) ? bk  : bv;
  _Float16* out       = (which == 0) ? qh  : (which == 1) ? kh  : vt;
  const bool tr = (which == 2);

  const int wave   = threadIdx.x >> 5;
  const int lane   = threadIdx.x & 31;
  const int lane15 = lane & 15;
  const int half   = lane >> 4;
  const int r0   = blockIdx.x * 16;   // global activation row in [0, B*S)
  const int col0 = wave * 128;        // output column base

  v8f acc[8] = {};
  for (int kk = 0; kk < Dm; kk += 32) {
    v16h a = load_tile_f32(x + (size_t)r0 * Dm + kk, Dm, lane);
#pragma unroll
    for (int t = 0; t < 8; ++t) {
      v16h bm = load_tile_f16(wt + (size_t)(col0 + t * 16) * Dm + kk, Dm, lane);
      acc[t] = wmma_f16(a, bm, acc[t]);
    }
  }

#pragma unroll
  for (int t = 0; t < 8; ++t) {
    const int c = col0 + t * 16 + lane15;   // global output column
    const int h = c >> 6, d = c & 63;
    const float bi = bias[c];
    if (!tr) {
#pragma unroll
      for (int i = 0; i < 8; ++i) {
        const int row = r0 + i + (half << 3);
        const int b = row >> 11, s = row & (Sq - 1);
        out[((size_t)(b * Hh + h) * Sq + s) * Dh + d] =
            (_Float16)(acc[t][i] + bi);
      }
    } else {
      // [b][h][d][S]: 8 consecutive s per lane -> one b128 store
      const int row = r0 + (half << 3);
      const int b = row >> 11, s = row & (Sq - 1);
      v8h pk;
#pragma unroll
      for (int i = 0; i < 8; ++i) pk[i] = (_Float16)(acc[t][i] + bi);
      *(v8h*)(out + ((size_t)(b * Hh + h) * Dh + d) * Sq + s) = pk;
    }
  }
}

// ---------------------------------------------------------------------------
// Kernel 3: flash attention. grid = (B*H, S/(QROWS*AWAVES)), block = 32*AWAVES.
// Each wave owns 32 q-rows (two 16-row groups sharing K/V tiles). All waves of
// a block share (b,h); K/V chunks (32 kpos) are staged into LDS once per block
// — by the Tensor Data Mover (two descriptors issued by wave 0, TENSORcnt)
// when available, else per-lane async global->LDS DMA — double-buffered so the
// next chunk's DMA overlaps the current chunk's WMMAs.
// Row-sums l_i are accumulated by WMMA against a ones-column B tile.
__global__ __launch_bounds__(32 * AWAVES) void attn_kernel(
    const _Float16* __restrict__ qh, const _Float16* __restrict__ kh,
    const _Float16* __restrict__ vt, const int* __restrict__ mask,
    float* __restrict__ out) {
  __shared__ _Float16 Kb[2][32 * 64];         // [kpos][d]   4 KB per buffer
  __shared__ _Float16 Vb[2][64 * 32];         // [d][kpos]   4 KB per buffer
  __shared__ _Float16 Pl[AWAVES][2][16 * 32]; // per-wave, per-rowgroup P staging

  const int bh = blockIdx.x;
  const int b = bh >> 4, h = bh & 15;
  const int tid    = threadIdx.x;
  const int wave   = tid >> 5;
  const int lane   = tid & 31;
  const int lane15 = lane & 15;
  const int half   = lane >> 4;
  const int q0 = (blockIdx.y * AWAVES + wave) * QROWS;

  const _Float16* Qh = qh + ((size_t)bh * Sq + q0) * Dh;
  const _Float16* Kh = kh + (size_t)bh * Sq * Dh;
  const _Float16* Vh = vt + (size_t)bh * Dh * Sq;
  const int* Mb = mask + ((size_t)b * Sq + q0) * Sq;

  // stage one 32-kpos K/V chunk into LDS buffer `buf`
  auto stage = [&](int k0, int buf) {
#if HAVE_TDM
    if (wave == 0) {
      // K tile: 32 rows x 64 elems, contiguous (stride = Dh)
      tdm_load_2d(Kh + (size_t)k0 * Dh, &Kb[buf][0], Dh, 32, Dh);
      // V^T tile: 64 rows x 32 elems, row stride = Sq elems
      tdm_load_2d(Vh + k0, &Vb[buf][0], 32, Dh, Sq);
    }
#else
    // K tile: 32 rows x 128B, contiguous 4 KB (one b128 per thread)
    copy16_g2l(Kh + (size_t)k0 * Dh + tid * 8, &Kb[buf][tid * 8]);
    // V^T tile: 64 rows x 64B (row stride S halfs)
    const int r = tid >> 2, o4 = (tid & 3) * 8;
    copy16_g2l(Vh + (size_t)r * Sq + k0 + o4, &Vb[buf][r * 32 + o4]);
#endif
  };
  auto stage_wait = [&]() {
#if HAVE_TDM
    if (wave == 0) __builtin_amdgcn_s_wait_tensorcnt(0);
#else
    wait_async_copies();
#endif
  };

  // Q A-tiles: [rowgroup][dchunk], loaded once, reused for all k.
  v16h aq[2][2];
#pragma unroll
  for (int rg = 0; rg < 2; ++rg) {
    aq[rg][0] = load_tile_f16(Qh + (size_t)rg * 16 * Dh, Dh, lane);
    aq[rg][1] = load_tile_f16(Qh + (size_t)rg * 16 * Dh + 32, Dh, lane);
  }

  // Ones B-tile: column 0 (lanes 0 and 16) holds 1.0 in every K slot, so
  // osum += P * bones accumulates per-row sums of P into C-layout column 0.
  v16h bones;
  {
    const _Float16 onev = (_Float16)((lane15 == 0) ? 1.0f : 0.0f);
#pragma unroll
    for (int j = 0; j < 16; ++j) bones[j] = onev;
  }

  v8f o[2][4] = {};
  v8f osum[2] = {};
  float m_i[2][8];
#pragma unroll
  for (int rg = 0; rg < 2; ++rg)
#pragma unroll
    for (int i = 0; i < 8; ++i) m_i[rg][i] = -1.0e30f;

  v16h pa[2];

  stage(0, 0);
  for (int k0 = 0, it = 0; k0 < Sq; k0 += 32, ++it) {
    const int buf = it & 1;
    stage_wait();            // our chunk-`it` DMAs done
    __syncthreads();         // everyone's DMAs visible; other buffer free
    if (k0 + 32 < Sq) stage(k0 + 32, buf ^ 1);
    __builtin_prefetch(Mb + (size_t)(half << 3) * Sq + k0 + 32, 0, 1);

    // K B-tiles from LDS, shared by both row-groups
    v16h b00 = load_tile_f16(&Kb[buf][0], Dh, lane);
    v16h b01 = load_tile_f16(&Kb[buf][32], Dh, lane);
    v16h b10 = load_tile_f16(&Kb[buf][16 * Dh], Dh, lane);
    v16h b11 = load_tile_f16(&Kb[buf][16 * Dh + 32], Dh, lane);

#pragma unroll
    for (int rg = 0; rg < 2; ++rg) {
      v8f s0 = {}, s1 = {};
      s0 = wmma_f16(aq[rg][0], b00, s0);
      s0 = wmma_f16(aq[rg][1], b01, s0);
      s1 = wmma_f16(aq[rg][0], b10, s1);
      s1 = wmma_f16(aq[rg][1], b11, s1);

      // online softmax: row stats per-VGPR; max-reduce via shfl_xor in halves
      float alpha[8];
#pragma unroll
      for (int i = 0; i < 8; ++i) {
        const int qrow = rg * 16 + i + (half << 3);
        const int* mrow = Mb + (size_t)qrow * Sq + k0;
        float sv0 = s0[i] * 0.125f;                 // 1/sqrt(64)
        float sv1 = s1[i] * 0.125f;
        sv0 = (mrow[lane15] == 0) ? NEGF : sv0;
        sv1 = (mrow[lane15 + 16] == 0) ? NEGF : sv1;
        float tm = fmaxf(sv0, sv1);
        tm = fmaxf(tm, __shfl_xor(tm, 1));
        tm = fmaxf(tm, __shfl_xor(tm, 2));
        tm = fmaxf(tm, __shfl_xor(tm, 4));
        tm = fmaxf(tm, __shfl_xor(tm, 8));
        const float mn = fmaxf(m_i[rg][i], tm);
        alpha[i] = __expf(m_i[rg][i] - mn);
        m_i[rg][i] = mn;
        const float p0 = __expf(sv0 - mn);
        const float p1 = __expf(sv1 - mn);
        const int r16 = i + (half << 3);
        Pl[wave][rg][r16 * 32 + lane15]      = (_Float16)p0;
        Pl[wave][rg][r16 * 32 + 16 + lane15] = (_Float16)p1;
      }
#pragma unroll
      for (int t = 0; t < 4; ++t)
#pragma unroll
        for (int i = 0; i < 8; ++i) o[rg][t][i] *= alpha[i];
#pragma unroll
      for (int i = 0; i < 8; ++i) osum[rg][i] *= alpha[i];

      // reload P in A-matrix layout (same-wave DS ops are in order)
      const _Float16* prow = &Pl[wave][rg][lane15 * 32 + (half << 3)];
#pragma unroll
      for (int j = 0; j < 8; ++j) {
        pa[rg][j] = prow[j];
        pa[rg][j + 8] = prow[j + 16];
      }
    }

    // O += P * V: V^T B-tiles loaded once, used by both row-groups
#pragma unroll
    for (int t = 0; t < 4; ++t) {
      v16h bvt = load_tile_f16(&Vb[buf][(t * 16) * 32], 32, lane);
      o[0][t] = wmma_f16(pa[0], bvt, o[0][t]);
      o[1][t] = wmma_f16(pa[1], bvt, o[1][t]);
    }
    // l += P * 1 (row sums via WMMA)
    osum[0] = wmma_f16(pa[0], bones, osum[0]);
    osum[1] = wmma_f16(pa[1], bones, osum[1]);
  }

#pragma unroll
  for (int rg = 0; rg < 2; ++rg) {
    float inv[8];
#pragma unroll
    for (int i = 0; i < 8; ++i) {
      const float l = __shfl(osum[rg][i], half << 4);  // broadcast column 0
      inv[i] = 1.0f / l;
    }
#pragma unroll
    for (int t = 0; t < 4; ++t) {
      const int d = h * Dh + t * 16 + lane15;
#pragma unroll
      for (int i = 0; i < 8; ++i) {
        const int s = q0 + rg * 16 + i + (half << 3);
        out[((size_t)b * Sq + s) * (Hh * Dh) + d] = o[rg][t][i] * inv[i];
      }
    }
  }
}

// ---------------------------------------------------------------------------
extern "C" void kernel_launch(void* const* d_in, const int* in_sizes, int n_in,
                              void* d_out, int out_size, void* d_ws,
                              size_t ws_size, hipStream_t stream) {
  (void)in_sizes; (void)n_in; (void)out_size; (void)ws_size;
  const float* q    = (const float*)d_in[0];
  const float* k    = (const float*)d_in[1];
  const float* v    = (const float*)d_in[2];
  const int*   msk  = (const int*)d_in[3];
  const float* wq   = (const float*)d_in[4];
  const float* bq   = (const float*)d_in[5];
  const float* wk   = (const float*)d_in[6];
  const float* bk   = (const float*)d_in[7];
  const float* wv   = (const float*)d_in[8];
  const float* bv   = (const float*)d_in[9];
  float* out = (float*)d_out;

  // workspace layout (f16): 3x W^T (2 MB each) + qh/kh/vt (16 MB each)
  _Float16* ws  = (_Float16*)d_ws;
  _Float16* wtq = ws;
  _Float16* wtk = wtq + (size_t)Dm * Dm;
  _Float16* wtv = wtk + (size_t)Dm * Dm;
  _Float16* qh  = wtv + (size_t)Dm * Dm;
  const size_t headElems = (size_t)Bsz * Hh * Sq * Dh;
  _Float16* kh = qh + headElems;
  _Float16* vt = kh + headElems;

  dim3 gT((Dm * Dm) / 256, 3);
  wtrans_kernel<<<gT, 256, 0, stream>>>(wq, wk, wv, wtq, wtk, wtv);

  dim3 gP((Bsz * Sq) / 16, 3);
  proj_kernel<<<gP, 256, 0, stream>>>(q, k, v, wtq, wtk, wtv, bq, bk, bv,
                                      qh, kh, vt);

  dim3 gA(Bsz * Hh, Sq / (QROWS * AWAVES));
  attn_kernel<<<gA, 32 * AWAVES, 0, stream>>>(qh, kh, vt, msk, out);
}
